// GraphSAGE_66864050864375
// MI455X (gfx1250) — matrix operand order repaired and checked by reading, not compiled, not run on hardware
//
#include <hip/hip_runtime.h>

#define NN 100000       // nodes
#define NE 1600000      // edges

typedef __attribute__((ext_vector_type(16))) __bf16        v16bf;
typedef __attribute__((ext_vector_type(8)))  float         v8f;
typedef __attribute__((ext_vector_type(4)))  unsigned int  v4u;

__device__ __forceinline__ unsigned short f2bf(float f) {
    unsigned int u = __float_as_uint(f);
    u += 0x7FFFu + ((u >> 16) & 1u);      // round-to-nearest-even
    return (unsigned short)(u >> 16);
}
__device__ __forceinline__ float bf2f(unsigned short s) {
    return __uint_as_float(((unsigned int)s) << 16);
}

// ---------------- elementwise / setup kernels ----------------

__global__ void k_cvt_bf16(const float* __restrict__ in, unsigned short* __restrict__ out, int n) {
    int i = blockIdx.x * blockDim.x + threadIdx.x;
    if (i < n) out[i] = f2bf(in[i]);
}

// W[din][dout] f32  ->  WT[dout][din] bf16
__global__ void k_wt_transpose(const float* __restrict__ W, unsigned short* __restrict__ WT,
                               int din, int dout) {
    int i = blockIdx.x * blockDim.x + threadIdx.x;
    if (i >= din * dout) return;
    int r = i / dout, c = i - r * dout;
    WT[c * din + r] = f2bf(W[i]);
}

__global__ void k_degree(const long long* __restrict__ dst, float* __restrict__ deg) {
    int e = blockIdx.x * blockDim.x + threadIdx.x;
    if (e < NE)
        __hip_atomic_fetch_add(&deg[(int)dst[e]], 1.0f,
                               __ATOMIC_RELAXED, __HIP_MEMORY_SCOPE_AGENT);
}

__global__ void k_inv_deg(float* deg) {
    int i = blockIdx.x * blockDim.x + threadIdx.x;
    if (i < NN) deg[i] = 1.0f / fmaxf(deg[i], 1.0f);
}

// edge-parallel scatter-sum: agg[dst] += bf2f(h[src]); 8 features per thread.
// h (bf16) and agg (f32) both fit in the 192MB L2 -> atomics are L2-resident.
template<int DIN>
__global__ void k_scatter(const unsigned short* __restrict__ h,
                          const long long* __restrict__ src,
                          const long long* __restrict__ dst,
                          float* __restrict__ agg) {
    constexpr int TPE = DIN / 8;
    int tid = blockIdx.x * blockDim.x + threadIdx.x;
    int e = tid / TPE;
    if (e >= NE) return;
    int c = tid - e * TPE;
    int s = (int)src[e], d = (int)dst[e];
    v4u q = *reinterpret_cast<const v4u*>(h + (size_t)s * DIN + c * 8);
    float* a = agg + (size_t)d * DIN + c * 8;
#pragma unroll
    for (int j = 0; j < 4; ++j) {
        unsigned int w = q[j];
        __hip_atomic_fetch_add(a + 2 * j,     bf2f((unsigned short)(w & 0xFFFFu)),
                               __ATOMIC_RELAXED, __HIP_MEMORY_SCOPE_AGENT);
        __hip_atomic_fetch_add(a + 2 * j + 1, bf2f((unsigned short)(w >> 16)),
                               __ATOMIC_RELAXED, __HIP_MEMORY_SCOPE_AGENT);
    }
}

// aggb = bf16(agg * invdeg[row]); DIN is a power of two
template<int DIN, int LOG2DIN>
__global__ void k_scale_bf16(const float* __restrict__ agg, const float* __restrict__ invdeg,
                             unsigned short* __restrict__ out) {
    int i = blockIdx.x * blockDim.x + threadIdx.x;
    if (i >= NN * DIN) return;
    out[i] = f2bf(agg[i] * invdeg[i >> LOG2DIN]);
}

// ---------------- fused SAGE GEMM: out = act(aggb@Wl + h@Wr + b) ----------------
// Block = 8 waves sharing ONE n-group (64 cols). Weight tiles for both matmuls are
// staged into LDS once per block via gfx1250 async-to-LDS DMA (ASYNCcnt), then each
// wave computes one 16-row M-tile x 4 N-tiles. Per K-step all 8 B fragments are
// loaded into distinct registers BEFORE the 8 WMMAs so ds_loads clause and the
// waits stagger instead of serializing each WMMA on a dscnt==0.
template<int DIN, int DOUT, bool RELU, bool OUT_BF16>
__global__ void __launch_bounds__(256)
k_sage_gemm(const unsigned short* __restrict__ Aagg,   // [NN, DIN] bf16 (mean-agg, pre-scaled)
            const unsigned short* __restrict__ Ah,     // [NN, DIN] bf16 (self features)
            const unsigned short* __restrict__ WlT,    // [DOUT, DIN] bf16
            const unsigned short* __restrict__ WrT,    // [DOUT, DIN] bf16
            const float* __restrict__ bias,            // [DOUT]
            unsigned short* __restrict__ outb,
            float* __restrict__ outf) {
    constexpr int NT        = 4;                 // 16x16 n-tiles per wave
    constexpr int TILE_COLS = 16 * NT;           // 64 cols per block
    constexpr int G         = DOUT / TILE_COLS;  // n-groups
    constexpr int WELEMS    = TILE_COLS * DIN;   // bf16 elems per weight tile
    constexpr int MT        = NN / 16;           // 6250 M-tiles

    // single LDS object -> placed at LDS offset 0 (byte offsets used by async DMA)
    __shared__ unsigned short smW[2 * WELEMS];   // [Wl tile | Wr tile], 32/64 KB

    const int mblk    = blockIdx.x / G;
    const int ng      = blockIdx.x - mblk * G;
    const int colBase = ng * TILE_COLS;
    const int wid     = threadIdx.x >> 5;
    const int lane    = threadIdx.x & 31;

    // ---- cooperative async fill of both weight tiles (contiguous rows of WT) ----
    {
        const unsigned short* gl = WlT + (size_t)colBase * DIN;
        const unsigned short* gr = WrT + (size_t)colBase * DIN;
        constexpr int CHUNKS = WELEMS / 8;       // 16B chunks per tile (divisible by 256)
#pragma unroll
        for (int c0 = 0; c0 < CHUNKS; c0 += 256) {
            const int c = c0 + threadIdx.x;
            unsigned off = (unsigned)(c * 16);
            asm volatile("global_load_async_to_lds_b128 %0, %1, off"
                         :: "v"(off), "v"(gl + c * 8) : "memory");
            asm volatile("global_load_async_to_lds_b128 %0, %1, off"
                         :: "v"(off + (unsigned)(2 * WELEMS)), "v"(gr + c * 8) : "memory");
        }
        asm volatile("s_wait_asynccnt 0" ::: "memory");
        __syncthreads();
    }

    const int mt = mblk * 8 + wid;
    if (mt < MT) {                               // wave-uniform: EXEC stays all-ones
        const int li = lane & 15;
        const int kg = lane >> 4;

        const unsigned short* aggRow = Aagg + (size_t)(mt * 16 + li) * DIN;
        const unsigned short* hRow   = Ah   + (size_t)(mt * 16 + li) * DIN;
        const unsigned short* sWl    = smW;
        const unsigned short* sWr    = smW + WELEMS;

        v8f acc[NT];
#pragma unroll
        for (int t = 0; t < NT; ++t)
#pragma unroll
            for (int j = 0; j < 8; ++j) acc[t][j] = 0.0f;

        union frag { v4u q[2]; v16bf v; };

#pragma unroll
        for (int k0 = 0; k0 < DIN; k0 += 32) {
            // A fragments (16x32 bf16, ISA 7.12.2): lane L, row=L&15;
            // halfs[0:7]=K[kg*8..], halfs[8:15]=K[16+kg*8..]
            frag fa, fh;
            fa.q[0] = *reinterpret_cast<const v4u*>(aggRow + k0 + kg * 8);
            fa.q[1] = *reinterpret_cast<const v4u*>(aggRow + k0 + 16 + kg * 8);
            fh.q[0] = *reinterpret_cast<const v4u*>(hRow   + k0 + kg * 8);
            fh.q[1] = *reinterpret_cast<const v4u*>(hRow   + k0 + 16 + kg * 8);

            // B fragments (32x16 bf16): lane L, col=L&15;
            // halfs[0:7]=K[kg*8..], halfs[8:15]=K[16+kg*8..] from WT[col]
            frag fbl[NT], fbr[NT];
#pragma unroll
            for (int t = 0; t < NT; ++t) {
                const int tcol = t * 16 + li;
                const unsigned short* bl = sWl + tcol * DIN + k0 + kg * 8;
                const unsigned short* br = sWr + tcol * DIN + k0 + kg * 8;
                fbl[t].q[0] = reinterpret_cast<const v4u*>(bl)[0];
                fbl[t].q[1] = reinterpret_cast<const v4u*>(bl)[2];
                fbr[t].q[0] = reinterpret_cast<const v4u*>(br)[0];
                fbr[t].q[1] = reinterpret_cast<const v4u*>(br)[2];
            }
#pragma unroll
            for (int t = 0; t < NT; ++t) {
                acc[t] = __builtin_amdgcn_wmma_f32_16x16x32_bf16(
                    false, fa.v, false, fbl[t].v, (short)0, acc[t], false, false);
                acc[t] = __builtin_amdgcn_wmma_f32_16x16x32_bf16(
                    false, fh.v, false, fbr[t].v, (short)0, acc[t], false, false);
            }
        }

        // Epilogue. C/D layout: VGPR j, lanes0-15 -> (M=j, N=lane); lanes16-31 -> (M=j+8, N=lane-16)
        const int mBase = mt * 16 + kg * 8;
#pragma unroll
        for (int t = 0; t < NT; ++t) {
            const int col = colBase + t * 16 + li;
            const float b = bias[col];
#pragma unroll
            for (int j = 0; j < 8; ++j) {
                float v = acc[t][j] + b;
                if (RELU) v = fmaxf(v, 0.0f);
                const size_t o = (size_t)(mBase + j) * DOUT + col;
                if (OUT_BF16) outb[o] = f2bf(v);
                else          outf[o] = v;
            }
        }
    }
}

// ---------------- host launcher ----------------

static inline int cdiv(long long a, int b) { return (int)((a + b - 1) / b); }

extern "C" void kernel_launch(void* const* d_in, const int* in_sizes, int n_in,
                              void* d_out, int out_size, void* d_ws, size_t ws_size,
                              hipStream_t stream) {
    (void)in_sizes; (void)n_in; (void)out_size; (void)ws_size;
    const float*     x   = (const float*)d_in[0];
    const long long* ei  = (const long long*)d_in[1];
    const float* Wl0 = (const float*)d_in[2];
    const float* b0  = (const float*)d_in[3];
    const float* Wr0 = (const float*)d_in[4];
    const float* Wl1 = (const float*)d_in[5];
    const float* b1  = (const float*)d_in[6];
    const float* Wr1 = (const float*)d_in[7];
    const float* Wl2 = (const float*)d_in[8];
    const float* b2  = (const float*)d_in[9];
    const float* Wr2 = (const float*)d_in[10];
    const long long* src = ei;
    const long long* dst = ei + NE;

    // workspace carve-out (~283 MB)
    char* ws = (char*)d_ws;
    size_t off = 0;
    auto carve = [&](size_t bytes) -> void* {
        void* p = ws + off;
        off += (bytes + 255) & ~(size_t)255;
        return p;
    };
    float*          deg  = (float*)         carve((size_t)NN * 4);
    float*          agg  = (float*)         carve((size_t)NN * 256 * 4);
    unsigned short* aggb = (unsigned short*)carve((size_t)NN * 256 * 2);
    unsigned short* h0   = (unsigned short*)carve((size_t)NN * 128 * 2);
    unsigned short* h1   = (unsigned short*)carve((size_t)NN * 256 * 2);
    unsigned short* h2   = (unsigned short*)carve((size_t)NN * 256 * 2);
    unsigned short* Wl0T = (unsigned short*)carve((size_t)128 * 256 * 2);
    unsigned short* Wr0T = (unsigned short*)carve((size_t)128 * 256 * 2);
    unsigned short* Wl1T = (unsigned short*)carve((size_t)256 * 256 * 2);
    unsigned short* Wr1T = (unsigned short*)carve((size_t)256 * 256 * 2);
    unsigned short* Wl2T = (unsigned short*)carve((size_t)256 * 128 * 2);
    unsigned short* Wr2T = (unsigned short*)carve((size_t)256 * 128 * 2);

    const int TB = 256;
    const int MBLKS = cdiv(NN / 16, 8);   // 782 m-blocks (8 M-tiles per block)

    // Conversions (re-run every call: deterministic, inputs never mutated)
    k_cvt_bf16<<<cdiv((long long)NN * 128, TB), TB, 0, stream>>>(x, h0, NN * 128);
    k_wt_transpose<<<cdiv(128 * 256, TB), TB, 0, stream>>>(Wl0, Wl0T, 128, 256);
    k_wt_transpose<<<cdiv(128 * 256, TB), TB, 0, stream>>>(Wr0, Wr0T, 128, 256);
    k_wt_transpose<<<cdiv(256 * 256, TB), TB, 0, stream>>>(Wl1, Wl1T, 256, 256);
    k_wt_transpose<<<cdiv(256 * 256, TB), TB, 0, stream>>>(Wr1, Wr1T, 256, 256);
    k_wt_transpose<<<cdiv(256 * 128, TB), TB, 0, stream>>>(Wl2, Wl2T, 256, 128);
    k_wt_transpose<<<cdiv(256 * 128, TB), TB, 0, stream>>>(Wr2, Wr2T, 256, 128);

    // Degrees (same graph for all layers)
    hipMemsetAsync(deg, 0, (size_t)NN * 4, stream);
    k_degree<<<cdiv(NE, TB), TB, 0, stream>>>(dst, deg);
    k_inv_deg<<<cdiv(NN, TB), TB, 0, stream>>>(deg);

    // ---- Layer 0: 128 -> 256, ReLU ----
    hipMemsetAsync(agg, 0, (size_t)NN * 128 * 4, stream);
    k_scatter<128><<<cdiv((long long)NE * 16, TB), TB, 0, stream>>>(h0, src, dst, agg);
    k_scale_bf16<128, 7><<<cdiv((long long)NN * 128, TB), TB, 0, stream>>>(agg, deg, aggb);
    k_sage_gemm<128, 256, true, true><<<MBLKS * 4, TB, 0, stream>>>(
        aggb, h0, Wl0T, Wr0T, b0, h1, nullptr);

    // ---- Layer 1: 256 -> 256, ReLU ----
    hipMemsetAsync(agg, 0, (size_t)NN * 256 * 4, stream);
    k_scatter<256><<<cdiv((long long)NE * 32, TB), TB, 0, stream>>>(h1, src, dst, agg);
    k_scale_bf16<256, 8><<<cdiv((long long)NN * 256, TB), TB, 0, stream>>>(agg, deg, aggb);
    k_sage_gemm<256, 256, true, true><<<MBLKS * 4, TB, 0, stream>>>(
        aggb, h1, Wl1T, Wr1T, b1, h2, nullptr);

    // ---- Layer 2: 256 -> 128, no activation, f32 output ----
    hipMemsetAsync(agg, 0, (size_t)NN * 256 * 4, stream);
    k_scatter<256><<<cdiv((long long)NE * 32, TB), TB, 0, stream>>>(h2, src, dst, agg);
    k_scale_bf16<256, 8><<<cdiv((long long)NN * 256, TB), TB, 0, stream>>>(agg, deg, aggb);
    k_sage_gemm<256, 128, false, false><<<MBLKS * 2, TB, 0, stream>>>(
        aggb, h2, Wl2T, Wr2T, b2, nullptr, (float*)d_out);
}
